// ST_GNN_Layer_52347061403820
// MI455X (gfx1250) — matrix operand order repaired and checked by reading, not compiled, not run on hardware
//
#include <hip/hip_runtime.h>

// ST-GNN layer (GATv2-style attention + GRUCell + LayerNorm) for MI455X/gfx1250.
// GEMMs use v_wmma_f32_16x16x32_bf16 (wave32, fp32 accum). Edge softmax uses
// L2-resident gather/scatter with float atomics (xh + agg tables are 51MB each,
// both fit the 192MB L2). N=100000 (divisible by 16), E=1600000 (div by 8).
// f32->bf16 conversion is fully vectorized (v_cvt_pk_bf16_f32 pairs) via
// __builtin_convertvector + shufflevector so no half-register fixup movs are
// needed on the WMMA issue path.

#define IN_CH   128
#define HID     128
#define HEADS   4
#define EDGE_DIM 16

typedef __attribute__((ext_vector_type(16))) __bf16         v16bf;
typedef __attribute__((ext_vector_type(8)))  __bf16         v8bf;
typedef __attribute__((ext_vector_type(8)))  float          v8f;
typedef __attribute__((ext_vector_type(16))) unsigned short v16us;

__device__ __forceinline__ unsigned short f32_to_bf16_u(float f) {
  return __builtin_bit_cast(unsigned short, (__bf16)f);
}
// build a 16-half A/B fragment from two 8-float runs (pair-aligned -> cvt_pk)
__device__ __forceinline__ v16bf pack_frag(v8f lo, v8f hi) {
  v8bf blo = __builtin_convertvector(lo, v8bf);
  v8bf bhi = __builtin_convertvector(hi, v8bf);
  return __builtin_shufflevector(blo, bhi, 0, 1, 2, 3, 4, 5, 6, 7,
                                           8, 9, 10, 11, 12, 13, 14, 15);
}
// monotone float <-> uint mapping for atomicMax-based softmax max
__device__ __forceinline__ unsigned flip_f(float f) {
  unsigned u = __builtin_bit_cast(unsigned, f);
  return (u & 0x80000000u) ? ~u : (u | 0x80000000u);
}
__device__ __forceinline__ float unflip_f(unsigned u) {
  unsigned v = (u & 0x80000000u) ? (u & 0x7FFFFFFFu) : ~u;
  return __builtin_bit_cast(float, v);
}
__device__ __forceinline__ float sigm(float x) { return 1.0f / (1.0f + __expf(-x)); }

// ---------------------------------------------------------------- weight pack
__global__ void k_cvt(const float* __restrict__ lin_w, const float* __restrict__ w_ih,
                      const float* __restrict__ w_hh, unsigned short* __restrict__ lw,
                      unsigned short* __restrict__ wi, unsigned short* __restrict__ wh) {
  int i = blockIdx.x * blockDim.x + threadIdx.x;
  if (i < 128 * 128) lw[i] = f32_to_bf16_u(lin_w[i]);
  if (i < 384 * 128) { wi[i] = f32_to_bf16_u(w_ih[i]); wh[i] = f32_to_bf16_u(w_hh[i]); }
}

// ---------------------------------------------------------------- zero init
__global__ void k_zero(float* __restrict__ agg, size_t na,
                       float* __restrict__ ssum, size_t ns,
                       unsigned* __restrict__ mmax, size_t nm) {
  size_t stride = (size_t)gridDim.x * blockDim.x;
  for (size_t i = blockIdx.x * (size_t)blockDim.x + threadIdx.x; i < na; i += stride) agg[i] = 0.0f;
  for (size_t i = blockIdx.x * (size_t)blockDim.x + threadIdx.x; i < ns; i += stride) ssum[i] = 0.0f;
  for (size_t i = blockIdx.x * (size_t)blockDim.x + threadIdx.x; i < nm; i += stride) mmax[i] = 0u;
}

// ---------------------------------------------------------------- xh = x @ lin_w^T
// block = 128 threads (4 waves); 16 rows x 128 cols per block; each wave: two
// 16x16 tiles, K=128 in four 32-chunks.  A layout (16-bit 16x32): lane m=lane&15,
// halves 0..15 -> K = koff+0..15, koff = (lane>=16)*16.
// B layout: lane n=lane&15, halves 0..15 -> K = koff2+0..15, koff2 = (lane>=16)*16.
__global__ __launch_bounds__(128)
void k_gemm_xh(const float* __restrict__ x, const unsigned short* __restrict__ wbf,
               float* __restrict__ xh) {
  const int lane = threadIdx.x & 31;
  const int wave = threadIdx.x >> 5;
  const int rowBase = blockIdx.x * 16;
  const int mrow = rowBase + (lane & 15);
  const int koff = (lane >> 4) * 16;

  v16bf afrag[4];
  const float* xrow = x + (size_t)mrow * IN_CH;
#pragma unroll
  for (int kc = 0; kc < 4; ++kc) {
    const float* p = xrow + kc * 32 + koff;
    afrag[kc] = pack_frag(*(const v8f*)(p), *(const v8f*)(p + 8));
  }
#pragma unroll
  for (int t = 0; t < 2; ++t) {
    const int tn = wave * 2 + t;                    // 0..7 -> cols tn*16..tn*16+15
    const int ncol = tn * 16 + (lane & 15);
    const unsigned short* wr = wbf + (size_t)ncol * IN_CH + (lane >> 4) * 16;
    v8f acc = {};
#pragma unroll
    for (int kc = 0; kc < 4; ++kc) {
      v16bf bfr = __builtin_bit_cast(v16bf, *(const v16us*)(wr + kc * 32));
      acc = __builtin_amdgcn_wmma_f32_16x16x32_bf16(false, afrag[kc], false, bfr,
                                                    (short)0, acc, false, false);
    }
    float* orow = xh + (size_t)(rowBase + (lane >> 4) * 8) * HID + tn * 16 + (lane & 15);
#pragma unroll
    for (int r = 0; r < 8; ++r) orow[(size_t)r * HID] = acc[r];
  }
}

// ---------------------------------------------------------------- per-node attention sums
// a_src[n,h] = sum_c xh[n,h*32+c]*att_src[h,c]; one block per node, one wave per head.
__global__ __launch_bounds__(128)
void k_node_att(const float* __restrict__ xh, const float* __restrict__ att_src,
                const float* __restrict__ att_dst, float* __restrict__ a_src,
                float* __restrict__ a_dst) {
  const int n = blockIdx.x;
  const int c = threadIdx.x;          // 0..127, head = c>>5
  float v = xh[(size_t)n * HID + c];
  float sa = v * att_src[c];
  float sd = v * att_dst[c];
#pragma unroll
  for (int off = 16; off > 0; off >>= 1) {
    sa += __shfl_xor(sa, off, 32);
    sd += __shfl_xor(sd, off, 32);
  }
  if ((c & 31) == 0) {
    a_src[n * HEADS + (c >> 5)] = sa;
    a_dst[n * HEADS + (c >> 5)] = sd;
  }
}

// ---------------------------------------------------------------- edge logits + segment max
__global__ __launch_bounds__(256)
void k_edge_logits(const int* __restrict__ src, const int* __restrict__ dst,
                   const float* __restrict__ a_src, const float* __restrict__ a_dst,
                   const float* __restrict__ edge_attr, const float* __restrict__ att_edge_w,
                   float* __restrict__ alpha, unsigned* __restrict__ mmax, int E) {
  __shared__ float w[HEADS * EDGE_DIM];
  if (threadIdx.x < HEADS * EDGE_DIM) w[threadIdx.x] = att_edge_w[threadIdx.x];
  __syncthreads();
  int e = blockIdx.x * blockDim.x + threadIdx.x;
  if (e >= E) return;
  int s = src[e], d = dst[e];
  float ea[EDGE_DIM];
#pragma unroll
  for (int k = 0; k < EDGE_DIM; ++k) ea[k] = edge_attr[(size_t)e * EDGE_DIM + k];
#pragma unroll
  for (int h = 0; h < HEADS; ++h) {
    float dot = 0.0f;
#pragma unroll
    for (int k = 0; k < EDGE_DIM; ++k) dot += ea[k] * w[h * EDGE_DIM + k];
    float al = a_src[s * HEADS + h] + a_dst[d * HEADS + h] + dot;
    al = (al >= 0.0f) ? al : 0.2f * al;            // leaky_relu(0.2)
    alpha[(size_t)e * HEADS + h] = al;
    atomicMax(&mmax[d * HEADS + h], flip_f(al));
  }
}

// ---------------------------------------------------------------- exp + segment sum
__global__ __launch_bounds__(256)
void k_edge_exp(const int* __restrict__ dst, float* __restrict__ alpha,
                const unsigned* __restrict__ mmax, float* __restrict__ ssum, int E) {
  int e = blockIdx.x * blockDim.x + threadIdx.x;
  if (e >= E) return;
  int d = dst[e];
#pragma unroll
  for (int h = 0; h < HEADS; ++h) {
    float al = alpha[(size_t)e * HEADS + h];
    float mv = unflip_f(mmax[d * HEADS + h]);
    float ev = __expf(al - mv);
    alpha[(size_t)e * HEADS + h] = ev;
    atomicAdd(&ssum[d * HEADS + h], ev);
  }
}

// ---------------------------------------------------------------- weighted aggregation
// one wave per edge; lane l handles channels l+32j (head j). xh/agg are L2-resident.
__global__ __launch_bounds__(256)
void k_aggregate(const int* __restrict__ src, const int* __restrict__ dst,
                 const float* __restrict__ xh, const float* __restrict__ alpha,
                 const float* __restrict__ ssum, float* __restrict__ agg, int E) {
  const int lane = threadIdx.x & 31;
  const int e = blockIdx.x * 8 + (threadIdx.x >> 5);
  if (e >= E) return;
  const int s = src[e], d = dst[e];
  const float* xr = xh + (size_t)s * HID;
  float* ar = agg + (size_t)d * HID;
#pragma unroll
  for (int j = 0; j < HEADS; ++j) {
    float anorm = alpha[(size_t)e * HEADS + j] / (ssum[d * HEADS + j] + 1e-16f);
    int c = j * 32 + lane;
    atomicAdd(&ar[c], xr[c] * anorm);
  }
}

// ---------------------------------------------------------------- GRU GEMMs + gates + LN
// block = 256 threads (8 waves), 16 rows. Waves 0-3 compute gi = (agg+bias)@w_ih^T
// (384 cols, 6 tiles each), waves 4-7 compute gh = h_prev@w_hh^T. Results in LDS,
// then gates + LayerNorm (one wave per 2 rows, lane reduction for mean/var).
__global__ __launch_bounds__(256)
void k_gru_ln(const float* __restrict__ agg, const float* __restrict__ gat_bias,
              const float* __restrict__ h_prev,
              const unsigned short* __restrict__ wih_bf, const unsigned short* __restrict__ whh_bf,
              const float* __restrict__ b_ih, const float* __restrict__ b_hh,
              const float* __restrict__ ln_g, const float* __restrict__ ln_b,
              float* __restrict__ y_out, float* __restrict__ h_out) {
  __shared__ float lds[16 * 768];                  // [row][gi 0..383 | gh 384..767]
  const int lane = threadIdx.x & 31;
  const int wave = threadIdx.x >> 5;
  const int rowBase = blockIdx.x * 16;
  const int mrow = rowBase + (lane & 15);
  const int koff = (lane >> 4) * 16;
  const bool is_gh = wave >= 4;
  const int wsub = is_gh ? (wave - 4) : wave;

  v16bf afrag[4];
  if (!is_gh) {
    const float* arow = agg + (size_t)mrow * HID;
#pragma unroll
    for (int kc = 0; kc < 4; ++kc) {
      const float* p = arow + kc * 32 + koff;
      const float* bb = gat_bias + kc * 32 + koff;
      v8f lo = *(const v8f*)(p) + *(const v8f*)(bb);
      v8f hi = *(const v8f*)(p + 8) + *(const v8f*)(bb + 8);
      afrag[kc] = pack_frag(lo, hi);
    }
  } else {
    const float* arow = h_prev + (size_t)mrow * HID;
#pragma unroll
    for (int kc = 0; kc < 4; ++kc) {
      const float* p = arow + kc * 32 + koff;
      afrag[kc] = pack_frag(*(const v8f*)(p), *(const v8f*)(p + 8));
    }
  }
  const unsigned short* W = is_gh ? whh_bf : wih_bf;
  float* ldsbase = lds + (is_gh ? 384 : 0);
#pragma unroll
  for (int t = 0; t < 6; ++t) {
    const int tn = wsub * 6 + t;                   // 0..23 -> cols tn*16..
    const int ncol = tn * 16 + (lane & 15);
    const unsigned short* wr = W + (size_t)ncol * HID + (lane >> 4) * 16;
    v8f acc = {};
#pragma unroll
    for (int kc = 0; kc < 4; ++kc) {
      v16bf bfr = __builtin_bit_cast(v16bf, *(const v16us*)(wr + kc * 32));
      acc = __builtin_amdgcn_wmma_f32_16x16x32_bf16(false, afrag[kc], false, bfr,
                                                    (short)0, acc, false, false);
    }
#pragma unroll
    for (int r = 0; r < 8; ++r)
      ldsbase[(r + (lane >> 4) * 8) * 768 + tn * 16 + (lane & 15)] = acc[r];
  }
  __syncthreads();

#pragma unroll
  for (int rr = 0; rr < 2; ++rr) {
    const int r = wave * 2 + rr;
    const int row = rowBase + r;
    float hv[4], sum = 0.0f, sumsq = 0.0f;
#pragma unroll
    for (int j = 0; j < 4; ++j) {
      const int c = j * 32 + lane;
      float gir = lds[r * 768 + c]             + b_ih[c];
      float giz = lds[r * 768 + 128 + c]       + b_ih[128 + c];
      float gin = lds[r * 768 + 256 + c]       + b_ih[256 + c];
      float ghr = lds[r * 768 + 384 + c]       + b_hh[c];
      float ghz = lds[r * 768 + 384 + 128 + c] + b_hh[128 + c];
      float ghn = lds[r * 768 + 384 + 256 + c] + b_hh[256 + c];
      float rg = sigm(gir + ghr);
      float zg = sigm(giz + ghz);
      float ng = tanhf(gin + rg * ghn);
      float hp = h_prev[(size_t)row * HID + c];
      float h = (1.0f - zg) * ng + zg * hp;
      hv[j] = h; sum += h; sumsq += h * h;
    }
#pragma unroll
    for (int off = 16; off > 0; off >>= 1) {
      sum   += __shfl_xor(sum, off, 32);
      sumsq += __shfl_xor(sumsq, off, 32);
    }
    const float mean = sum * (1.0f / HID);
    const float var  = sumsq * (1.0f / HID) - mean * mean;
    const float rstd = rsqrtf(var + 1e-5f);
#pragma unroll
    for (int j = 0; j < 4; ++j) {
      const int c = j * 32 + lane;
      y_out[(size_t)row * HID + c] = (hv[j] - mean) * rstd * ln_g[c] + ln_b[c];
      h_out[(size_t)row * HID + c] = hv[j];
    }
  }
}

// ================================================================ launcher
extern "C" void kernel_launch(void* const* d_in, const int* in_sizes, int n_in,
                              void* d_out, int out_size, void* d_ws, size_t ws_size,
                              hipStream_t stream) {
  const float* x          = (const float*)d_in[0];
  const float* edge_attr  = (const float*)d_in[1];
  const float* h_prev     = (const float*)d_in[2];
  const float* lin_w      = (const float*)d_in[3];
  const float* att_src    = (const float*)d_in[4];
  const float* att_dst    = (const float*)d_in[5];
  const float* att_edge_w = (const float*)d_in[6];
  const float* gat_bias   = (const float*)d_in[7];
  const float* w_ih       = (const float*)d_in[8];
  const float* w_hh       = (const float*)d_in[9];
  const float* b_ih       = (const float*)d_in[10];
  const float* b_hh       = (const float*)d_in[11];
  const float* ln_g       = (const float*)d_in[12];
  const float* ln_b       = (const float*)d_in[13];
  const int*   edge_index = (const int*)d_in[14];

  const int N = in_sizes[0] / IN_CH;      // 100000
  const int E = in_sizes[14] / 2;         // 1600000
  const int* src = edge_index;
  const int* dst = edge_index + E;

  // workspace carve-up (~134 MB)
  char* wsb = (char*)d_ws;
  size_t off = 0;
  auto carve = [&](size_t bytes) -> char* {
    char* p = wsb + off;
    off += (bytes + 255) & ~(size_t)255;
    return p;
  };
  float*          xh     = (float*)carve((size_t)N * HID * 4);
  float*          agg    = (float*)carve((size_t)N * HID * 4);
  float*          a_srcb = (float*)carve((size_t)N * HEADS * 4);
  float*          a_dstb = (float*)carve((size_t)N * HEADS * 4);
  float*          ssum   = (float*)carve((size_t)N * HEADS * 4);
  unsigned*       mmax   = (unsigned*)carve((size_t)N * HEADS * 4);
  float*          alphab = (float*)carve((size_t)E * HEADS * 4);
  unsigned short* lw_bf  = (unsigned short*)carve(128 * 128 * 2);
  unsigned short* wi_bf  = (unsigned short*)carve(384 * 128 * 2);
  unsigned short* wh_bf  = (unsigned short*)carve(384 * 128 * 2);

  k_cvt<<<(384 * 128 + 255) / 256, 256, 0, stream>>>(lin_w, w_ih, w_hh, lw_bf, wi_bf, wh_bf);
  k_zero<<<2048, 256, 0, stream>>>(agg, (size_t)N * HID, ssum, (size_t)N * HEADS,
                                   mmax, (size_t)N * HEADS);
  k_gemm_xh<<<N / 16, 128, 0, stream>>>(x, lw_bf, xh);
  k_node_att<<<N, 128, 0, stream>>>(xh, att_src, att_dst, a_srcb, a_dstb);
  k_edge_logits<<<(E + 255) / 256, 256, 0, stream>>>(src, dst, a_srcb, a_dstb, edge_attr,
                                                     att_edge_w, alphab, mmax, E);
  k_edge_exp<<<(E + 255) / 256, 256, 0, stream>>>(dst, alphab, mmax, ssum, E);
  k_aggregate<<<(E + 7) / 8, 256, 0, stream>>>(src, dst, xh, alphab, ssum, agg, E);

  float* y_out = (float*)d_out;
  float* h_out = y_out + (size_t)N * HID;
  k_gru_ln<<<N / 16, 256, 0, stream>>>(agg, gat_bias, h_prev, wi_bf, wh_bf,
                                       b_ih, b_hh, ln_g, ln_b, y_out, h_out);
}
